// MoE_Grouped_Token_86414741995922
// MI455X (gfx1250) — matrix operand order repaired and checked by reading, not compiled, not run on hardware
//
#include <hip/hip_runtime.h>
#include <hip/hip_bf16.h>
#include <math.h>

// Problem constants (B=1): T=2048 tokens, D=1024, E=64, H=256, G=64, K=16
#define T_TOK 2048
#define DIM   1024
#define NE    64
#define NH    256
#define NG    64
#define NTQ   32            // T / G
#define KSEL  16
#define BUDGET (NTQ * KSEL) // 512 selected (group, expert) slots

typedef __attribute__((ext_vector_type(16))) __bf16 v16bf;
typedef __attribute__((ext_vector_type(8)))  float  v8f;

union ABFrag {
    v16bf v;
    unsigned short u[16];
    uint4 q[2];
};

__device__ __forceinline__ unsigned short f2bf(float f) {
    union { float f; unsigned u; } v; v.f = f;
    unsigned r = v.u + 0x7FFFu + ((v.u >> 16) & 1u);   // round-to-nearest-even
    return (unsigned short)(r >> 16);
}

// ---------------------------------------------------------------------------
// 1) Router: logits = x @ Wr + br, softmax over E.  One block per token.
// ---------------------------------------------------------------------------
__global__ void router_kernel(const float* __restrict__ x,
                              const float* __restrict__ Wr,
                              const float* __restrict__ br,
                              float* __restrict__ P,
                              float* __restrict__ fs) {
    int t = blockIdx.x;
    int e = threadIdx.x;                 // 64 threads
    const float* xr = x + t * DIM;
    float acc = br[e];
    for (int d = 0; d < DIM; ++d)
        acc = fmaf(xr[d], Wr[d * NE + e], acc);
    __shared__ float sm[NE];
    sm[e] = acc;
    __syncthreads();
    float mx = -1e30f;
    for (int i = 0; i < NE; ++i) mx = fmaxf(mx, sm[i]);
    float ex = __expf(acc - mx);
    __syncthreads();
    sm[e] = ex;
    __syncthreads();
    float s = 0.f;
    for (int i = 0; i < NE; ++i) s += sm[i];
    P[t * NE + e] = ex / s;
    fs[t * NE + e] = 0.f;
}

// ---------------------------------------------------------------------------
// Shared bitonic descending sort (key,f32 / idx,i32) of n elements in LDS.
// ---------------------------------------------------------------------------
__device__ void bitonic_desc(float* key, int* idx, int n, int tid, int nthr) {
    for (int k = 2; k <= n; k <<= 1) {
        for (int j = k >> 1; j > 0; j >>= 1) {
            __syncthreads();
            for (int i = tid; i < n; i += nthr) {
                int ixj = i ^ j;
                if (ixj > i) {
                    bool sw = ((i & k) == 0) ? (key[i] < key[ixj])
                                             : (key[i] > key[ixj]);
                    if (sw) {
                        float tk = key[i]; key[i] = key[ixj]; key[ixj] = tk;
                        int   ti = idx[i]; idx[i] = idx[ixj]; idx[ixj] = ti;
                    }
                }
            }
        }
    }
    __syncthreads();
}

// ---------------------------------------------------------------------------
// 2) Per-expert sort of token scores (desc) + group sums. One block per expert.
// ---------------------------------------------------------------------------
__global__ void sort_kernel(const float* __restrict__ P,
                            int* __restrict__ order,
                            float* __restrict__ gscore) {
    __shared__ float key[T_TOK];
    __shared__ int   idx[T_TOK];
    int e = blockIdx.x, tid = threadIdx.x;
    for (int i = tid; i < T_TOK; i += blockDim.x) {
        key[i] = P[i * NE + e];
        idx[i] = i;
    }
    bitonic_desc(key, idx, T_TOK, tid, blockDim.x);
    for (int i = tid; i < T_TOK; i += blockDim.x)
        order[e * T_TOK + i] = idx[i];
    if (tid < NTQ) {
        float s = 0.f;
        for (int g = 0; g < NG; ++g) s += key[tid * NG + g];
        gscore[tid * NE + e] = s;
    }
}

// ---------------------------------------------------------------------------
// 3) Global top-512 of the 2048 (group, expert) scores. One block.
// ---------------------------------------------------------------------------
__global__ void topk_kernel(const float* __restrict__ gscore,
                            int* __restrict__ gmask) {
    __shared__ float key[NTQ * NE];
    __shared__ int   idx[NTQ * NE];
    int tid = threadIdx.x;
    for (int i = tid; i < NTQ * NE; i += blockDim.x) {
        key[i] = gscore[i];
        idx[i] = i;
        gmask[i] = 0;
    }
    bitonic_desc(key, idx, NTQ * NE, tid, blockDim.x);
    for (int p = tid; p < BUDGET; p += blockDim.x)
        gmask[idx[p]] = 1;
}

// ---------------------------------------------------------------------------
// 4) Scatter mask through per-expert sorted order back to (token, expert).
// ---------------------------------------------------------------------------
__global__ void scatter_kernel(const float* __restrict__ P,
                               const int* __restrict__ order,
                               const int* __restrict__ gmask,
                               float* __restrict__ fs) {
    int g = blockIdx.x * blockDim.x + threadIdx.x;
    if (g >= NE * T_TOK) return;
    int e = g / T_TOK, p = g % T_TOK;
    int q = p >> 6;
    if (gmask[q * NE + e]) {
        int t = order[e * T_TOK + p];
        fs[t * NE + e] = P[t * NE + e];
    }
}

// ---------------------------------------------------------------------------
// 5) Per-token renormalize fs.
// ---------------------------------------------------------------------------
__global__ void norm_kernel(float* __restrict__ fs) {
    int t = blockIdx.x, e = threadIdx.x;
    __shared__ float sm[NE];
    float v = fs[t * NE + e];
    sm[e] = v;
    __syncthreads();
    float s = 0.f;
    for (int i = 0; i < NE; ++i) s += sm[i];
    fs[t * NE + e] = v / (s + 1e-6f);
}

// ---------------------------------------------------------------------------
// 6) Compact per-expert active token lists (counts are multiples of 64).
// ---------------------------------------------------------------------------
__global__ void compact_kernel(const int* __restrict__ gmask,
                               const int* __restrict__ order,
                               const float* __restrict__ fs,
                               int* __restrict__ list,
                               float* __restrict__ lfs,
                               int* __restrict__ counts) {
    int e = blockIdx.x, tid = threadIdx.x;   // 64 threads
    int base = 0;                            // uniform across block
    for (int q = 0; q < NTQ; ++q) {
        if (gmask[q * NE + e]) {
            int t = order[e * T_TOK + q * NG + tid];
            list[e * T_TOK + base + tid] = t;
            lfs [e * T_TOK + base + tid] = fs[t * NE + e];
            base += NG;
        }
    }
    if (tid == 0) counts[e] = base;
}

// ---------------------------------------------------------------------------
// 7) Zero output (atomically accumulated afterwards).
// ---------------------------------------------------------------------------
__global__ void zero_kernel(float* __restrict__ out) {
    int i = blockIdx.x * blockDim.x + threadIdx.x;
    float4 z = {0.f, 0.f, 0.f, 0.f};
    ((float4*)out)[i] = z;
}

// ---------------------------------------------------------------------------
// 8) Weight repack: fp32 [e][K][N] -> bf16 fragment-major WMMA-B layout.
// For each 32x16 (KxN) tile, lane L's 16 halves (col n0+(L&15),
// rows k0+(L<16?0:16)..+15) are contiguous: 32B per lane, coalesced.
// ---------------------------------------------------------------------------
__global__ void repack_kernel(const float* __restrict__ W,
                              unsigned short* __restrict__ Wp,
                              int K, int N) {
    int e   = blockIdx.y;
    int idx = blockIdx.x * blockDim.x + threadIdx.x;  // (kt*nNt + nt)*32 + lane
    int nKt = K >> 5, nNt = N >> 4;
    if (idx >= nKt * nNt * 32) return;
    int lane = idx & 31;
    int tile = idx >> 5;
    int nt = tile % nNt, kt = tile / nNt;
    int n  = nt * 16 + (lane & 15);
    int kb = kt * 32 + ((lane < 16) ? 0 : 16);
    const float* p = W + (size_t)e * K * N + (size_t)kb * N + n;
    ABFrag f;
#pragma unroll
    for (int kk = 0; kk < 16; ++kk)
        f.u[kk] = f2bf(p[(size_t)kk * N]);            // coalesced across lanes
    unsigned short* q = Wp + ((size_t)e * nKt * nNt * 32 + (size_t)idx) * 16;
    *(uint4*)(q)     = f.q[0];
    *(uint4*)(q + 8) = f.q[1];
}

// ---------------------------------------------------------------------------
// WMMA fragment loaders (wave32; contiguous k order: lane half h holds k = 16h..16h+15 in A and B).
// ---------------------------------------------------------------------------
__device__ __forceinline__ v16bf load_a_frag(const unsigned short* base, int ld,
                                             int k0, int lane) {
    int m    = lane & 15;
    int koff = (lane < 16) ? 0 : 16;
    ABFrag a;
    const unsigned short* p = base + m * ld + k0 + koff;
    a.q[0] = *(const uint4*)(p);
    a.q[1] = *(const uint4*)(p + 8);
    return a.v;
}

__device__ __forceinline__ v16bf load_b_frag_packed(const unsigned short* __restrict__ Wp,
                                                    int nNt, int kt, int nt, int lane) {
    const unsigned short* p = Wp + ((size_t)(((kt * nNt + nt) << 5) + lane) << 4);
    ABFrag b;
    b.q[0] = *(const uint4*)(p);       // global_load_b128
    b.q[1] = *(const uint4*)(p + 8);   // global_load_b128
    return b.v;
}

__device__ __forceinline__ v16bf load_b_frag_f32(const float* __restrict__ W,
                                                 int ld, int k0, int n0, int lane) {
    int n  = n0 + (lane & 15);
    int kb = k0 + ((lane < 16) ? 0 : 16);
    ABFrag b;
    const float* p = W + (size_t)kb * ld + n;
#pragma unroll
    for (int kk = 0; kk < 16; ++kk)
        b.u[kk] = f2bf(p[(size_t)kk * ld]);
    return b.v;
}

// ---------------------------------------------------------------------------
// 9) Expert FFN. 32-token M-tile per block (2 x 16 sub-tiles), 8 wave32.
//    up/gate: wave owns 2 of 16 H-tiles; down: wave owns 8 of 64 D-tiles,
//    processed in two passes of 4 to cap live accumulators at 64 VGPRs.
// ---------------------------------------------------------------------------
template <bool PACKED>
__global__ void __launch_bounds__(256)
ffn_kernel(const float* __restrict__ x,
           const float* __restrict__ Wup,   const float* __restrict__ bup,
           const float* __restrict__ Wgate, const float* __restrict__ bgate,
           const float* __restrict__ Wdown, const float* __restrict__ bdown,
           const unsigned short* __restrict__ WupP,
           const unsigned short* __restrict__ WgateP,
           const unsigned short* __restrict__ WdownP,
           const int* __restrict__ list,    const float* __restrict__ lfs,
           const int* __restrict__ counts,  float* __restrict__ out) {
    const int e    = blockIdx.y;
    const int tile = blockIdx.x;                   // 32-token tile
    if (tile * 32 >= counts[e]) return;            // uniform per block

    const int XLD = DIM + 8;                       // 1032 shorts (16B-aligned)
    const int HLD = NH + 8;                        // 264 shorts
    __shared__ __align__(16) unsigned short Xs[32 * (DIM + 8)];   // ~64.5 KB
    __shared__ __align__(16) unsigned short Hs[32 * (NH + 8)];    // ~16.5 KB
    __shared__ int   toks[32];
    __shared__ float tf[32];

    const int tid  = threadIdx.x;
    const int lane = tid & 31;
    const int wave = tid >> 5;

    if (tid < 32) {
        int slot  = e * T_TOK + tile * 32 + tid;
        toks[tid] = list[slot];
        tf[tid]   = lfs[slot];
    }
    __syncthreads();

    // ---- load 32 x 1024 activation tile -> bf16 LDS ----
    {
        int m  = tid >> 3;                          // 32 rows, 8 threads/row
        int st = tid & 7;
        const float4* xr = (const float4*)(x + (size_t)toks[m] * DIM);
        unsigned short* dst = &Xs[m * XLD];
        for (int d4 = st; d4 < DIM / 4; d4 += 8) {
            float4 v = xr[d4];
            dst[d4 * 4 + 0] = f2bf(v.x);
            dst[d4 * 4 + 1] = f2bf(v.y);
            dst[d4 * 4 + 2] = f2bf(v.z);
            dst[d4 * 4 + 3] = f2bf(v.w);
        }
    }
    __syncthreads();

    // ---- up & gate: (32 x 1024) @ (1024 x 256) ----
    const int nNtH = NH >> 4;                      // 16 N-tiles over H
    const float*          WuF = Wup   + (size_t)e * DIM * NH;
    const float*          WgF = Wgate + (size_t)e * DIM * NH;
    const unsigned short* WuP = WupP   + (size_t)e * (DIM >> 5) * nNtH * 32 * 16;
    const unsigned short* WgP = WgateP + (size_t)e * (DIM >> 5) * nNtH * 32 * 16;
    const int nt0 = wave * 2;
    v8f accu[2][2] = {}, accg[2][2] = {};          // [m-subtile][n-tile]
    for (int kt = 0; kt < DIM / 32; ++kt) {
        v16bf a0 = load_a_frag(Xs,            XLD, kt * 32, lane);
        v16bf a1 = load_a_frag(Xs + 16 * XLD, XLD, kt * 32, lane);
        if (PACKED && kt + 1 < DIM / 32) {
            __builtin_prefetch(WuP + ((size_t)(((kt + 1) * nNtH + nt0) * 32 + lane) << 4), 0, 1);
            __builtin_prefetch(WgP + ((size_t)(((kt + 1) * nNtH + nt0) * 32 + lane) << 4), 0, 1);
        }
#pragma unroll
        for (int ni = 0; ni < 2; ++ni) {
            v16bf bu = PACKED ? load_b_frag_packed(WuP, nNtH, kt, nt0 + ni, lane)
                              : load_b_frag_f32(WuF, NH, kt * 32, (nt0 + ni) * 16, lane);
            v16bf bg = PACKED ? load_b_frag_packed(WgP, nNtH, kt, nt0 + ni, lane)
                              : load_b_frag_f32(WgF, NH, kt * 32, (nt0 + ni) * 16, lane);
            accu[0][ni] = __builtin_amdgcn_wmma_f32_16x16x32_bf16(false, a0, false, bu, (short)0, accu[0][ni], false, false);
            accu[1][ni] = __builtin_amdgcn_wmma_f32_16x16x32_bf16(false, a1, false, bu, (short)0, accu[1][ni], false, false);
            accg[0][ni] = __builtin_amdgcn_wmma_f32_16x16x32_bf16(false, a0, false, bg, (short)0, accg[0][ni], false, false);
            accg[1][ni] = __builtin_amdgcn_wmma_f32_16x16x32_bf16(false, a1, false, bg, (short)0, accg[1][ni], false, false);
        }
    }

    // ---- h = silu(up + bup) * (gate + bgate) * fs -> bf16 LDS (32 x 256) ----
    {
        const int mbase = (lane < 16) ? 0 : 8;
#pragma unroll
        for (int ni = 0; ni < 2; ++ni) {
            int   col = (nt0 + ni) * 16 + (lane & 15);
            float buv = bup  [e * NH + col];
            float bgv = bgate[e * NH + col];
#pragma unroll
            for (int m2 = 0; m2 < 2; ++m2) {
#pragma unroll
                for (int r = 0; r < 8; ++r) {
                    int   m = m2 * 16 + mbase + r;
                    float u = accu[m2][ni][r] + buv;
                    float g = accg[m2][ni][r] + bgv;
                    float h = (u / (1.f + __expf(-u))) * g * tf[m];
                    Hs[m * HLD + col] = f2bf(h);
                }
            }
        }
    }
    __syncthreads();

    // ---- down: (32 x 256) @ (256 x 1024), two passes of 4 N-tiles/wave ----
    const int nNtD = DIM >> 4;                     // 64 N-tiles over D
    const float*          WdF = Wdown  + (size_t)e * NH * DIM;
    const unsigned short* WdP = WdownP + (size_t)e * (NH >> 5) * nNtD * 32 * 16;
    const int   mbase = (lane < 16) ? 0 : 8;
    const float* bd = bdown + e * DIM;

#pragma unroll 1
    for (int half = 0; half < 2; ++half) {
        v8f zero = {};
        v8f accd[2][4];
#pragma unroll
        for (int m2 = 0; m2 < 2; ++m2)
#pragma unroll
            for (int j = 0; j < 4; ++j) accd[m2][j] = zero;

        for (int kt = 0; kt < NH / 32; ++kt) {
            v16bf a0 = load_a_frag(Hs,            HLD, kt * 32, lane);
            v16bf a1 = load_a_frag(Hs + 16 * HLD, HLD, kt * 32, lane);
#pragma unroll
            for (int j = 0; j < 4; ++j) {
                int nt = wave * 8 + half * 4 + j;
                v16bf b = PACKED ? load_b_frag_packed(WdP, nNtD, kt, nt, lane)
                                 : load_b_frag_f32(WdF, DIM, kt * 32, nt * 16, lane);
                accd[0][j] = __builtin_amdgcn_wmma_f32_16x16x32_bf16(false, a0, false, b, (short)0, accd[0][j], false, false);
                accd[1][j] = __builtin_amdgcn_wmma_f32_16x16x32_bf16(false, a1, false, b, (short)0, accd[1][j], false, false);
            }
        }

        // write out: += down + fs * bdown (hw f32 atomics; tokens shared)
#pragma unroll
        for (int j = 0; j < 4; ++j) {
            int   col = (wave * 8 + half * 4 + j) * 16 + (lane & 15);
            float bdv = bd[col];
#pragma unroll
            for (int m2 = 0; m2 < 2; ++m2) {
#pragma unroll
                for (int r = 0; r < 8; ++r) {
                    int   m   = m2 * 16 + mbase + r;
                    float val = accd[m2][j][r] + tf[m] * bdv;
                    unsafeAtomicAdd(&out[(size_t)toks[m] * DIM + col], val);
                }
            }
        }
    }
}

// ---------------------------------------------------------------------------
static inline size_t align256(size_t v) { return (v + 255) & ~(size_t)255; }

extern "C" void kernel_launch(void* const* d_in, const int* in_sizes, int n_in,
                              void* d_out, int out_size, void* d_ws, size_t ws_size,
                              hipStream_t stream) {
    (void)in_sizes; (void)n_in; (void)out_size;

    const float* x     = (const float*)d_in[0];
    const float* Wr    = (const float*)d_in[1];
    const float* br    = (const float*)d_in[2];
    const float* Wup   = (const float*)d_in[3];
    const float* bup   = (const float*)d_in[4];
    const float* Wgate = (const float*)d_in[5];
    const float* bgate = (const float*)d_in[6];
    const float* Wdown = (const float*)d_in[7];
    const float* bdown = (const float*)d_in[8];
    float* out = (float*)d_out;

    char*  base = (char*)d_ws;
    size_t off  = 0;
    float* P      = (float*)(base + off); off += align256((size_t)T_TOK * NE * 4);
    float* fs     = (float*)(base + off); off += align256((size_t)T_TOK * NE * 4);
    int*   order  = (int*)  (base + off); off += align256((size_t)NE * T_TOK * 4);
    float* gscore = (float*)(base + off); off += align256((size_t)NTQ * NE * 4);
    int*   gmask  = (int*)  (base + off); off += align256((size_t)NTQ * NE * 4);
    int*   list   = (int*)  (base + off); off += align256((size_t)NE * T_TOK * 4);
    float* lfs    = (float*)(base + off); off += align256((size_t)NE * T_TOK * 4);
    int*   counts = (int*)  (base + off); off += align256((size_t)NE * 4);

    const size_t wElems = (size_t)NE * DIM * NH;   // 16.78M per weight tensor
    unsigned short* WupP   = (unsigned short*)(base + off); size_t offP = off + align256(wElems * 2);
    unsigned short* WgateP = (unsigned short*)(base + offP); offP += align256(wElems * 2);
    unsigned short* WdownP = (unsigned short*)(base + offP); offP += align256(wElems * 2);
    const bool packed = (ws_size >= offP);

    router_kernel <<<T_TOK, NE, 0, stream>>>(x, Wr, br, P, fs);
    sort_kernel   <<<NE, 256, 0, stream>>>(P, order, gscore);
    topk_kernel   <<<1, 256, 0, stream>>>(gscore, gmask);
    scatter_kernel<<<(NE * T_TOK) / 256, 256, 0, stream>>>(P, order, gmask, fs);
    norm_kernel   <<<T_TOK, NE, 0, stream>>>(fs);
    compact_kernel<<<NE, NG, 0, stream>>>(gmask, order, fs, list, lfs, counts);
    zero_kernel   <<<(T_TOK * DIM) / (256 * 4), 256, 0, stream>>>(out);

    dim3 rg((DIM / 32) * (NH / 16) * 32 / 256, NE);   // 64 x 64 (up/gate shape)
    if (packed) {
        repack_kernel<<<rg, 256, 0, stream>>>(Wup,   WupP,   DIM, NH);
        repack_kernel<<<rg, 256, 0, stream>>>(Wgate, WgateP, DIM, NH);
        dim3 rgd((NH / 32) * (DIM / 16) * 32 / 256, NE);
        repack_kernel<<<rgd, 256, 0, stream>>>(Wdown, WdownP, NH, DIM);
    }

    dim3 g(T_TOK / 32, NE);                       // 64 max 32-token tiles x 64 experts
    if (packed) {
        ffn_kernel<true><<<g, 256, 0, stream>>>(x, Wup, bup, Wgate, bgate, Wdown, bdown,
                                                WupP, WgateP, WdownP,
                                                list, lfs, counts, out);
    } else {
        ffn_kernel<false><<<g, 256, 0, stream>>>(x, Wup, bup, Wgate, bgate, Wdown, bdown,
                                                 WupP, WgateP, WdownP,
                                                 list, lfs, counts, out);
    }
}
